// LSTMPINN_86088324481430
// MI455X (gfx1250) — matrix operand order repaired and checked by reading, not compiled, not run on hardware
//
#include <hip/hip_runtime.h>
#include <hip/hip_bf16.h>
#include <math.h>

typedef _Float16 v16h __attribute__((ext_vector_type(16)));
typedef _Float16 v8h  __attribute__((ext_vector_type(8)));
typedef float    v8f  __attribute__((ext_vector_type(8)));

#define WS 72     // weight row stride in halfs (144B: 16B aligned, conflict-free)
#define HS 104    // h/fused staging row stride in halfs (208B: 16B aligned, conflict-free)
#define NWAVES 8
#define TSTEPS 9

union Frag { v16h v; v8h h[2]; };

__device__ __forceinline__ v8f wmma16(v16h a, v16h b, v8f c) {
  return __builtin_amdgcn_wmma_f32_16x16x32_f16(false, a, false, b, (short)0, c, false, false);
}

// A fragment (16x32 f16): lanes 0-15 rows M=0..15; element e -> K = (e<8 ? e : 16+(e-8)) + 8*half + 32*kc
__device__ __forceinline__ v16h loadA(const _Float16* base, int kc) {
  int lane = threadIdx.x & 31;
  int m = lane & 15, half = lane >> 4;
  const _Float16* row = base + m * HS + kc * 32 + 8 * half;
  Frag f;
  f.h[0] = *(const v8h*)(row);       // K = 32kc + 8*half .. +7
  f.h[1] = *(const v8h*)(row + 16);  // K = 32kc + 16 + 8*half .. +7
  return f.v;
}

// B fragment (32x16 f16): lane n = l%16 selects column (gate); element e -> K = (e<8 ? e : 16+(e-8)) + 8*half + 32*kc
__device__ __forceinline__ v16h loadB(const _Float16* base, int stride, int nt, int kc) {
  int lane = threadIdx.x & 31;
  int n = lane & 15, half = lane >> 4;
  const _Float16* row = base + (nt * 16 + n) * stride + kc * 32 + 8 * half;
  Frag f;
  f.h[0] = *(const v8h*)(row);
  f.h[1] = *(const v8h*)(row + 16);
  return f.v;
}

// C/D tile (16x16 f32): VGPR v holds row M = v + 8*half, lane n = l%16 is column.
__device__ __forceinline__ void storeTile(_Float16* base, int col0, v8f c) {
  int lane = threadIdx.x & 31;
  int n = lane & 15, half = lane >> 4;
#pragma unroll
  for (int v = 0; v < 8; ++v)
    base[(v + 8 * half) * HS + col0 + n] = (_Float16)c[v];
}

__device__ __forceinline__ float sigm(float x)   { return 1.f / (1.f + __expf(-x)); }
__device__ __forceinline__ float tanh_f(float x) { float e = __expf(2.f * x); return 1.f - 2.f / (e + 1.f); }

__device__ void stageW(_Float16* dst, const float* src, int rows, int cols, int stride, int srcRows) {
  int total = rows * stride;
  for (int i = threadIdx.x; i < total; i += blockDim.x) {
    int r = i / stride, c = i - r * stride;
    float v = (c < cols && r < srcRows) ? src[r * cols + c] : 0.f;
    dst[i] = (_Float16)v;
  }
}

__global__ void __launch_bounds__(256, 1)
lstm_pinn_wmma(const float* __restrict__ xy,
               const float* __restrict__ w_ih0, const float* __restrict__ w_hh0, const float* __restrict__ b0,
               const float* __restrict__ w_ih1, const float* __restrict__ w_hh1, const float* __restrict__ b1,
               const float* __restrict__ fus_w, const float* __restrict__ fus_b,
               const float* __restrict__ res_w1, const float* __restrict__ res_b1,
               const float* __restrict__ res_w2, const float* __restrict__ res_b2,
               const float* __restrict__ head_w1, const float* __restrict__ head_b1,
               const float* __restrict__ head_w2, const float* __restrict__ head_b2,
               float* __restrict__ out, int N)
{
  extern __shared__ _Float16 smem[];
  _Float16* sWhh0 = smem;                       // 256 x WS
  _Float16* sWih1 = sWhh0 + 256 * WS;           // 256 x WS
  _Float16* sWhh1 = sWih1 + 256 * WS;           // 256 x WS
  _Float16* sFus  = sWhh1 + 256 * WS;           // 64 x HS (K padded 66->96)
  _Float16* sRes1 = sFus  + 64 * HS;            // 3 x 64 x WS
  _Float16* sRes2 = sRes1 + 3 * 64 * WS;        // 3 x 64 x WS
  _Float16* sHw1  = sRes2 + 3 * 64 * WS;        // 64 x WS
  _Float16* sHw2  = sHw1  + 64 * WS;            // 16 x WS (rows padded 5->16)
  _Float16* sHbuf = sHw2  + 16 * WS;            // NWAVES x 2 x 16 x HS
  float*    sSeq  = (float*)(sHbuf + NWAVES * 2 * 16 * HS); // NWAVES x 16 x 12

  stageW(sWhh0, w_hh0, 256, 64, WS, 256);
  stageW(sWih1, w_ih1, 256, 64, WS, 256);
  stageW(sWhh1, w_hh1, 256, 64, WS, 256);
  stageW(sFus,  fus_w,  64, 66, HS, 64);
  for (int k = 0; k < 3; ++k) {
    stageW(sRes1 + k * 64 * WS, res_w1 + k * 64 * 64, 64, 64, WS, 64);
    stageW(sRes2 + k * 64 * WS, res_w2 + k * 64 * 64, 64, 64, WS, 64);
  }
  stageW(sHw1, head_w1, 64, 64, WS, 64);
  stageW(sHw2, head_w2, 16, 64, WS, 5);
  __syncthreads();

  int lane = threadIdx.x & 31;
  int wid  = threadIdx.x >> 5;
  int n = lane & 15, half = lane >> 4;
  _Float16* hA = sHbuf + wid * (2 * 16 * HS);
  _Float16* hB = hA + 16 * HS;
  float* seq = sSeq + wid * (16 * 12);

  int gwave = blockIdx.x * NWAVES + wid;
  int nwaves = gridDim.x * NWAVES;
  int numTiles = (N + 15) >> 4;

#pragma unroll 1
  for (int tile = gwave; tile < numTiles; tile += nwaves) {
    long base = (long)tile * 16;

    // zero both staging buffers (h0prev = h1prev = 0, and fused K-pad zeros)
    v8h zer;
#pragma unroll
    for (int e = 0; e < 8; ++e) zer[e] = (_Float16)0.f;
    for (int i = lane; i < 16 * (HS / 8); i += 32) {
      ((v8h*)hA)[i] = zer;
      ((v8h*)hB)[i] = zer;
    }

    // build the 9 scalar sequence features per sample
    if (lane < 16) {
      long r = base + lane; if (r >= N) r = N - 1;
      float x = xy[2 * r], y = xy[2 * r + 1];
      const float PI = 3.14159265358979323846f;
      float* s = seq + lane * 12;
      s[0] = x; s[1] = y; s[2] = x * x; s[3] = y * y; s[4] = x * y;
      s[5] = __sinf(PI * x); s[6] = __cosf(PI * x);
      s[7] = __sinf(PI * y); s[8] = __cosf(PI * y);
    }

    v8f c0s[4], c1s[4];
#pragma unroll
    for (int ft = 0; ft < 4; ++ft)
#pragma unroll
      for (int v = 0; v < 8; ++v) { c0s[ft][v] = 0.f; c1s[ft][v] = 0.f; }

#pragma unroll 1
    for (int t = 0; t < TSTEPS; ++t) {
      float sv[8];
#pragma unroll
      for (int v = 0; v < 8; ++v) sv[v] = seq[(v + 8 * half) * 12 + t];

      // ---- layer 0: G = pre0 + h0_prev @ Whh0^T ----
      v16h a0 = loadA(hA, 0), a1 = loadA(hA, 1);
      v8f h0n[4];
#pragma unroll
      for (int ft = 0; ft < 4; ++ft) {
        v8f g[4];
#pragma unroll
        for (int gate = 0; gate < 4; ++gate) {
          int nt = gate * 4 + ft, g0 = nt * 16;
          float wi = w_ih0[g0 + n], bb = b0[g0 + n];
          v8f c;
#pragma unroll
          for (int v = 0; v < 8; ++v) c[v] = wi * sv[v] + bb;
          c = wmma16(a0, loadB(sWhh0, WS, nt, 0), c);
          c = wmma16(a1, loadB(sWhh0, WS, nt, 1), c);
          g[gate] = c;
        }
#pragma unroll
        for (int v = 0; v < 8; ++v) {
          float ig = sigm(g[0][v]), fg = sigm(g[1][v]);
          float gg = tanh_f(g[2][v]), og = sigm(g[3][v]);
          float cc = fg * c0s[ft][v] + ig * gg;
          c0s[ft][v] = cc;
          h0n[ft][v] = og * tanh_f(cc);
        }
      }
#pragma unroll
      for (int ft = 0; ft < 4; ++ft) storeTile(hA, ft * 16, h0n[ft]);

      // ---- layer 1: G = b1 + h0_t @ Wih1^T + h1_prev @ Whh1^T ----
      v16h x0 = loadA(hA, 0), x1 = loadA(hA, 1);
      v16h p0 = loadA(hB, 0), p1 = loadA(hB, 1);
      v8f h1n[4];
#pragma unroll
      for (int ft = 0; ft < 4; ++ft) {
        v8f g[4];
#pragma unroll
        for (int gate = 0; gate < 4; ++gate) {
          int nt = gate * 4 + ft, g0 = nt * 16;
          float bb = b1[g0 + n];
          v8f c;
#pragma unroll
          for (int v = 0; v < 8; ++v) c[v] = bb;
          c = wmma16(x0, loadB(sWih1, WS, nt, 0), c);
          c = wmma16(x1, loadB(sWih1, WS, nt, 1), c);
          c = wmma16(p0, loadB(sWhh1, WS, nt, 0), c);
          c = wmma16(p1, loadB(sWhh1, WS, nt, 1), c);
          g[gate] = c;
        }
#pragma unroll
        for (int v = 0; v < 8; ++v) {
          float ig = sigm(g[0][v]), fg = sigm(g[1][v]);
          float gg = tanh_f(g[2][v]), og = sigm(g[3][v]);
          float cc = fg * c1s[ft][v] + ig * gg;
          c1s[ft][v] = cc;
          h1n[ft][v] = og * tanh_f(cc);
        }
      }
#pragma unroll
      for (int ft = 0; ft < 4; ++ft) storeTile(hB, ft * 16, h1n[ft]);
    }

    // ---- fused = [h1_last, x, y] (K padded to 96); z = tanh(fused @ fus_w^T + fus_b) ----
    if (lane < 16) {
      hB[lane * HS + 64] = (_Float16)seq[lane * 12 + 0];
      hB[lane * HS + 65] = (_Float16)seq[lane * 12 + 1];
    }
    v16h f0 = loadA(hB, 0), f1 = loadA(hB, 1), f2 = loadA(hB, 2);
    v8f z[4];
#pragma unroll
    for (int nt = 0; nt < 4; ++nt) {
      float bb = fus_b[nt * 16 + n];
      v8f c;
#pragma unroll
      for (int v = 0; v < 8; ++v) c[v] = bb;
      c = wmma16(f0, loadB(sFus, HS, nt, 0), c);
      c = wmma16(f1, loadB(sFus, HS, nt, 1), c);
      c = wmma16(f2, loadB(sFus, HS, nt, 2), c);
#pragma unroll
      for (int v = 0; v < 8; ++v) z[nt][v] = tanh_f(c[v]);
    }

    // ---- residual blocks ----
#pragma unroll 1
    for (int k = 0; k < 3; ++k) {
#pragma unroll
      for (int nt = 0; nt < 4; ++nt) storeTile(hA, nt * 16, z[nt]);
      v16h za = loadA(hA, 0), zb = loadA(hA, 1);
      v8f tt[4];
#pragma unroll
      for (int nt = 0; nt < 4; ++nt) {
        float bb = res_b1[k * 64 + nt * 16 + n];
        v8f c;
#pragma unroll
        for (int v = 0; v < 8; ++v) c[v] = bb;
        c = wmma16(za, loadB(sRes1 + k * 64 * WS, WS, nt, 0), c);
        c = wmma16(zb, loadB(sRes1 + k * 64 * WS, WS, nt, 1), c);
#pragma unroll
        for (int v = 0; v < 8; ++v) tt[nt][v] = tanh_f(c[v]);
      }
#pragma unroll
      for (int nt = 0; nt < 4; ++nt) storeTile(hB, nt * 16, tt[nt]);
      v16h ta = loadA(hB, 0), tb = loadA(hB, 1);
#pragma unroll
      for (int nt = 0; nt < 4; ++nt) {
        float bb = res_b2[k * 64 + nt * 16 + n];
        v8f c;
#pragma unroll
        for (int v = 0; v < 8; ++v) c[v] = bb;
        c = wmma16(ta, loadB(sRes2 + k * 64 * WS, WS, nt, 0), c);
        c = wmma16(tb, loadB(sRes2 + k * 64 * WS, WS, nt, 1), c);
#pragma unroll
        for (int v = 0; v < 8; ++v) z[nt][v] += tanh_f(c[v]);
      }
    }

    // ---- head ----
#pragma unroll
    for (int nt = 0; nt < 4; ++nt) storeTile(hA, nt * 16, z[nt]);
    v16h ha0 = loadA(hA, 0), ha1 = loadA(hA, 1);
    v8f hh[4];
#pragma unroll
    for (int nt = 0; nt < 4; ++nt) {
      float bb = head_b1[nt * 16 + n];
      v8f c;
#pragma unroll
      for (int v = 0; v < 8; ++v) c[v] = bb;
      c = wmma16(ha0, loadB(sHw1, WS, nt, 0), c);
      c = wmma16(ha1, loadB(sHw1, WS, nt, 1), c);
#pragma unroll
      for (int v = 0; v < 8; ++v) hh[nt][v] = tanh_f(c[v]);
    }
#pragma unroll
    for (int nt = 0; nt < 4; ++nt) storeTile(hB, nt * 16, hh[nt]);
    v16h hb0 = loadA(hB, 0), hb1 = loadA(hB, 1);
    float bb = 0.f;
    if (n < 5) bb = head_b2[n];
    v8f c;
#pragma unroll
    for (int v = 0; v < 8; ++v) c[v] = bb;
    c = wmma16(hb0, loadB(sHw2, WS, 0, 0), c);
    c = wmma16(hb1, loadB(sHw2, WS, 0, 1), c);
#pragma unroll
    for (int v = 0; v < 8; ++v) {
      long r = base + v + 8 * half;
      if (r < (long)N && n < 5) out[r * 5 + n] = c[v];
    }
  }
}

extern "C" void kernel_launch(void* const* d_in, const int* in_sizes, int n_in,
                              void* d_out, int out_size, void* d_ws, size_t ws_size,
                              hipStream_t stream) {
  (void)n_in; (void)out_size; (void)d_ws; (void)ws_size;
  const float* xy      = (const float*)d_in[0];
  const float* w_ih0   = (const float*)d_in[1];
  const float* w_hh0   = (const float*)d_in[2];
  const float* b0      = (const float*)d_in[3];
  const float* w_ih1   = (const float*)d_in[4];
  const float* w_hh1   = (const float*)d_in[5];
  const float* b1      = (const float*)d_in[6];
  const float* fus_w   = (const float*)d_in[7];
  const float* fus_b   = (const float*)d_in[8];
  const float* res_w1  = (const float*)d_in[9];
  const float* res_b1  = (const float*)d_in[10];
  const float* res_w2  = (const float*)d_in[11];
  const float* res_b2  = (const float*)d_in[12];
  const float* head_w1 = (const float*)d_in[13];
  const float* head_b1 = (const float*)d_in[14];
  const float* head_w2 = (const float*)d_in[15];
  const float* head_b2 = (const float*)d_in[16];
  int N = in_sizes[0] / 2;

  size_t halfs = (size_t)3 * 256 * WS + 64 * HS + 6 * 64 * WS + 64 * WS + 16 * WS
               + (size_t)NWAVES * 2 * 16 * HS;
  size_t smemBytes = halfs * 2 + (size_t)NWAVES * 16 * 12 * 4;

  (void)hipFuncSetAttribute((const void*)lstm_pinn_wmma,
                            hipFuncAttributeMaxDynamicSharedMemorySize,
                            (int)smemBytes);

  dim3 grid(256), block(256);
  hipLaunchKernelGGL(lstm_pinn_wmma, grid, block, smemBytes, stream,
                     xy, w_ih0, w_hh0, b0, w_ih1, w_hh1, b1,
                     fus_w, fus_b, res_w1, res_b1, res_w2, res_b2,
                     head_w1, head_b1, head_w2, head_b2,
                     (float*)d_out, N);
}
